// GATBlock_63350767616009
// MI455X (gfx1250) — matrix operand order repaired and checked by reading, not compiled, not run on hardware
//
#include <hip/hip_runtime.h>

// ---------------- types for WMMA ----------------
typedef __attribute__((ext_vector_type(16))) __bf16          v16bf;
typedef __attribute__((ext_vector_type(16))) unsigned short  v16us;
typedef __attribute__((ext_vector_type(8)))  float           v8f;

#define HID   256
#define HEADS 4
#define DH    64
#define SLOPE 0.2f

// ---------------- helpers ----------------
__device__ __forceinline__ unsigned short bf16_rne(float f) {
    unsigned u = __float_as_uint(f);
    unsigned r = u + 0x7fffu + ((u >> 16) & 1u);   // round-to-nearest-even
    return (unsigned short)(r >> 16);
}
__device__ __forceinline__ float bf16f(unsigned short h) {
    return __uint_as_float(((unsigned)h) << 16);
}
// order-preserving float -> uint key for atomicMax over signed floats
__device__ __forceinline__ unsigned fkey(float f) {
    unsigned u = __float_as_uint(f);
    return (u & 0x80000000u) ? ~u : (u | 0x80000000u);
}
__device__ __forceinline__ float fdecode(unsigned k) {
    unsigned u = (k & 0x80000000u) ? (k & 0x7fffffffu) : ~k;
    return __uint_as_float(u);
}

// ---------------- 0) init: zero out / denom / max-keys ----------------
__global__ void gat_init(float* __restrict__ out, float* __restrict__ denom,
                         unsigned* __restrict__ emax_key, int N) {
    int t = blockIdx.x * 256 + threadIdx.x;
    if (t < N * HID) out[t] = 0.0f;
    if (t < N * HEADS) { denom[t] = 0.0f; emax_key[t] = 0u; }
}

// ---------------- 1) W -> transposed bf16 hi/lo ----------------
// W is [HID(K) x 256(Nout)] row-major; store Wt[n*256 + k] so B fragments are contiguous.
__global__ void gat_wprep(const float* __restrict__ W,
                          unsigned short* __restrict__ wt_hi,
                          unsigned short* __restrict__ wt_lo) {
    int t = blockIdx.x * 256 + threadIdx.x;   // 65536 total
    int k = t >> 8, n = t & 255;
    float f = W[t];
    unsigned short hi = bf16_rne(f);
    wt_hi[n * 256 + k] = hi;
    wt_lo[n * 256 + k] = bf16_rne(f - bf16f(hi));
}

// ---------------- 2) GEMM: h = x @ W  via bf16x2 WMMA, fp32 accumulate ----------------
// One wave computes a 16(M) x 64(N) tile; 4 waves per block cover N=256; grid.x = ceil(N/16).
__global__ void __launch_bounds__(128)
gat_gemm(const float* __restrict__ x,
         const unsigned short* __restrict__ wt_hi,
         const unsigned short* __restrict__ wt_lo,
         float* __restrict__ h, int N) {
    const int lane = threadIdx.x & 31;
    const int wave = threadIdx.x >> 5;          // 0..3 -> 64-col subtile
    const int m0   = blockIdx.x * 16;
    const int n0   = wave * 64;
    const int mrow = lane & 15;
    const int kh   = lane >> 4;                 // lane-half: 0/1

    int row_ld = m0 + mrow; if (row_ld > N - 1) row_ld = N - 1;   // clamp (EXEC must stay full)
    const float* xrow = x + (size_t)row_ld * HID;

    v8f c0 = {}, c1 = {}, c2 = {}, c3 = {};

    for (int k0 = 0; k0 < HID; k0 += 32) {
        // ---- A fragment: f32 -> bf16 hi/lo on the fly ----
        // A k order used here (same as B): per lane, elems 0..15 = K(kh*16+0..15) contiguous
        const float* ap = xrow + k0 + kh * 16;
        float af[16];
        #pragma unroll
        for (int j = 0; j < 8; j++) { af[j] = ap[j]; af[8 + j] = ap[8 + j]; }
        v16us ahi_u, alo_u;
        #pragma unroll
        for (int j = 0; j < 16; j++) {
            unsigned short hi = bf16_rne(af[j]);
            ahi_u[j] = hi;
            alo_u[j] = bf16_rne(af[j] - bf16f(hi));
        }
        v16bf ahi = __builtin_bit_cast(v16bf, ahi_u);
        v16bf alo = __builtin_bit_cast(v16bf, alo_u);

        #pragma unroll
        for (int ns = 0; ns < 4; ns++) {
            // B layout: lane holds column n, elems 0..15 = K(k0 + kh*16 + 0..15) contiguous
            const int ncol = n0 + ns * 16 + (lane & 15);
            const unsigned short* bh = wt_hi + (size_t)ncol * HID + k0 + kh * 16;
            const unsigned short* bl = wt_lo + (size_t)ncol * HID + k0 + kh * 16;
            v16us bhu, blu;
            *(uint4*)&bhu       = *(const uint4*)bh;
            *((uint4*)&bhu + 1) = *(const uint4*)(bh + 8);
            *(uint4*)&blu       = *(const uint4*)bl;
            *((uint4*)&blu + 1) = *(const uint4*)(bl + 8);
            v16bf bhi = __builtin_bit_cast(v16bf, bhu);
            v16bf blo = __builtin_bit_cast(v16bf, blu);

            v8f* cc = (ns == 0) ? &c0 : (ns == 1) ? &c1 : (ns == 2) ? &c2 : &c3;
            *cc = __builtin_amdgcn_wmma_f32_16x16x32_bf16(false, ahi, false, bhi, (short)0, *cc, false, false);
            *cc = __builtin_amdgcn_wmma_f32_16x16x32_bf16(false, ahi, false, blo, (short)0, *cc, false, false);
            *cc = __builtin_amdgcn_wmma_f32_16x16x32_bf16(false, alo, false, bhi, (short)0, *cc, false, false);
        }
    }

    // C/D layout: elem j -> row (kh*8 + j), col = lane&15 within 16-col subtile
    #pragma unroll
    for (int ns = 0; ns < 4; ns++) {
        v8f cc = (ns == 0) ? c0 : (ns == 1) ? c1 : (ns == 2) ? c2 : c3;
        const int col = n0 + ns * 16 + (lane & 15);
        #pragma unroll
        for (int j = 0; j < 8; j++) {
            int row = m0 + kh * 8 + j;
            if (row < N) h[(size_t)row * HID + col] = cc[j];
        }
    }
}

// ---------------- 3) per-node attention dots: a_s, a_d ----------------
// One wave per node: lane covers 8 contiguous dims; head = lane/8; shfl-reduce in 8-lane groups.
__global__ void __launch_bounds__(256)
gat_attdot(const float* __restrict__ h, const float* __restrict__ att_src,
           const float* __restrict__ att_dst, float* __restrict__ a_s,
           float* __restrict__ a_d, int N) {
    int w    = (blockIdx.x * 256 + threadIdx.x) >> 5;
    int lane = threadIdx.x & 31;
    if (w >= N) return;
    int base = lane * 8;            // flattened (head,dim): head = base/64
    int head = lane >> 3;
    const float* hp = h + (size_t)w * HID + base;
    const float* sp = att_src + base;     // att_src flat [HEADS*DH] == [256]
    const float* dp = att_dst + base;
    float ss = 0.f, sd = 0.f;
    #pragma unroll
    for (int j = 0; j < 8; j++) { float hv = hp[j]; ss += hv * sp[j]; sd += hv * dp[j]; }
    #pragma unroll
    for (int off = 1; off < 8; off <<= 1) {
        ss += __shfl_xor(ss, off, 32);
        sd += __shfl_xor(sd, off, 32);
    }
    if ((lane & 7) == 0) { a_s[w * HEADS + head] = ss; a_d[w * HEADS + head] = sd; }
}

// ---------------- 4) edge pass 1: leaky-relu logits + segment max ----------------
__global__ void gat_edge_logits(const int* __restrict__ ei, const float* __restrict__ a_s,
                                const float* __restrict__ a_d, float* __restrict__ elog,
                                unsigned* __restrict__ emax_key, int E, int Etot) {
    int e = blockIdx.x * 256 + threadIdx.x;
    if (e >= Etot) return;
    int src, dst;
    if (e < E) { src = ei[e]; dst = ei[E + e]; } else { src = dst = e - E; }
    float4 as = *(const float4*)(a_s + (size_t)src * 4);
    float4 ad = *(const float4*)(a_d + (size_t)dst * 4);
    float v[4] = {as.x + ad.x, as.y + ad.y, as.z + ad.z, as.w + ad.w};
    #pragma unroll
    for (int t = 0; t < 4; t++) v[t] = v[t] > 0.f ? v[t] : SLOPE * v[t];
    *(float4*)(elog + (size_t)e * 4) = make_float4(v[0], v[1], v[2], v[3]);
    unsigned* mk = emax_key + (size_t)dst * 4;
    #pragma unroll
    for (int t = 0; t < 4; t++) atomicMax(mk + t, fkey(v[t]));
}

// ---------------- 5) edge pass 2: ez = exp(e - max), denom += ez ----------------
__global__ void gat_edge_exp(const int* __restrict__ ei, float* __restrict__ elog,
                             const unsigned* __restrict__ emax_key, float* __restrict__ denom,
                             int E, int Etot) {
    int e = blockIdx.x * 256 + threadIdx.x;
    if (e >= Etot) return;
    int dst = (e < E) ? ei[E + e] : (e - E);
    float4 lv = *(const float4*)(elog + (size_t)e * 4);
    const unsigned* kk = emax_key + (size_t)dst * 4;
    float z0 = __expf(lv.x - fdecode(kk[0]));
    float z1 = __expf(lv.y - fdecode(kk[1]));
    float z2 = __expf(lv.z - fdecode(kk[2]));
    float z3 = __expf(lv.w - fdecode(kk[3]));
    *(float4*)(elog + (size_t)e * 4) = make_float4(z0, z1, z2, z3);
    float* dn = denom + (size_t)dst * 4;
    atomicAdd(dn + 0, z0); atomicAdd(dn + 1, z1);
    atomicAdd(dn + 2, z2); atomicAdd(dn + 3, z3);
}

// ---------------- 6) edge pass 3: out[dst] += h[src] * alpha ----------------
// One wave per edge; each lane covers 8 coalesced f32 of the 256-wide row.
__global__ void __launch_bounds__(256)
gat_edge_scatter(const int* __restrict__ ei, const float* __restrict__ elog,
                 const float* __restrict__ denom, const float* __restrict__ h,
                 float* __restrict__ out, int E, int Etot) {
    int w    = (blockIdx.x * 256 + threadIdx.x) >> 5;
    int lane = threadIdx.x & 31;
    if (w >= Etot) return;
    int src, dst;
    if (w < E) { src = ei[w]; dst = ei[E + w]; } else { src = dst = w - E; }
    float4 ez = *(const float4*)(elog + (size_t)w * 4);
    float4 dn = *(const float4*)(denom + (size_t)dst * 4);
    float al0 = ez.x / (dn.x + 1e-16f);
    float al1 = ez.y / (dn.y + 1e-16f);
    float al2 = ez.z / (dn.z + 1e-16f);
    float al3 = ez.w / (dn.w + 1e-16f);
    const float* hs = h + (size_t)src * HID;
    float* od = out + (size_t)dst * HID;
    #pragma unroll
    for (int i = 0; i < 8; i++) {                   // idx in [32i, 32i+32) -> head = i/2 (compile-time)
        int idx = i * 32 + lane;
        float a = (i < 2) ? al0 : (i < 4) ? al1 : (i < 6) ? al2 : al3;
        atomicAdd(od + idx, hs[idx] * a);
    }
}

// ---------------- 7) bias + relu ----------------
__global__ void gat_finish(float* __restrict__ out, const float* __restrict__ bias, int total) {
    int t = blockIdx.x * 256 + threadIdx.x;
    if (t >= total) return;
    float v = out[t] + bias[t & (HID - 1)];
    out[t] = v > 0.f ? v : 0.f;
}

// ---------------- launcher ----------------
extern "C" void kernel_launch(void* const* d_in, const int* in_sizes, int n_in,
                              void* d_out, int out_size, void* d_ws, size_t ws_size,
                              hipStream_t stream) {
    const float* x       = (const float*)d_in[0];
    const float* W       = (const float*)d_in[1];
    const float* att_src = (const float*)d_in[2];
    const float* att_dst = (const float*)d_in[3];
    const float* bias    = (const float*)d_in[4];
    const int*   ei      = (const int*)d_in[5];

    const int N    = in_sizes[0] / HID;
    const int E    = in_sizes[5] / 2;
    const int Etot = E + N;

    char* ws = (char*)d_ws;
    size_t off = 0;
    auto alloc = [&](size_t bytes) -> char* {
        char* p = ws + off;
        off = (off + bytes + 255) & ~(size_t)255;
        return p;
    };
    float*          h        = (float*)alloc((size_t)N * HID * 4);
    unsigned short* wt_hi    = (unsigned short*)alloc((size_t)HID * 256 * 2);
    unsigned short* wt_lo    = (unsigned short*)alloc((size_t)HID * 256 * 2);
    float*          a_s      = (float*)alloc((size_t)N * HEADS * 4);
    float*          a_d      = (float*)alloc((size_t)N * HEADS * 4);
    unsigned*       emax_key = (unsigned*)alloc((size_t)N * HEADS * 4);
    float*          denom    = (float*)alloc((size_t)N * HEADS * 4);
    float*          elog     = (float*)alloc((size_t)Etot * HEADS * 4);

    float* out = (float*)d_out;

    gat_init<<<dim3((N * HID + 255) / 256), dim3(256), 0, stream>>>(out, denom, emax_key, N);
    gat_wprep<<<dim3((HID * 256) / 256), dim3(256), 0, stream>>>(W, wt_hi, wt_lo);
    gat_gemm<<<dim3((N + 15) / 16), dim3(128), 0, stream>>>(x, wt_hi, wt_lo, h, N);
    gat_attdot<<<dim3((N * 32 + 255) / 256), dim3(256), 0, stream>>>(h, att_src, att_dst, a_s, a_d, N);
    gat_edge_logits<<<dim3((Etot + 255) / 256), dim3(256), 0, stream>>>(ei, a_s, a_d, elog, emax_key, E, Etot);
    gat_edge_exp<<<dim3((Etot + 255) / 256), dim3(256), 0, stream>>>(ei, elog, emax_key, denom, E, Etot);
    gat_edge_scatter<<<dim3((Etot * 32 + 255) / 256), dim3(256), 0, stream>>>(ei, elog, denom, h, out, E, Etot);
    gat_finish<<<dim3((N * HID + 255) / 256), dim3(256), 0, stream>>>(out, bias, N * HID);
}
